// UnderstandingEncoder_7215545057646
// MI455X (gfx1250) — matrix operand order repaired and checked by reading, not compiled, not run on hardware
//
#include <hip/hip_runtime.h>

// ---------------------------------------------------------------------------
// Types for CDNA5 WMMA (wave32): v16bf A/B operands, v8f accumulator.
// ---------------------------------------------------------------------------
typedef __bf16 bf16;
typedef __attribute__((ext_vector_type(2)))  __bf16 bf16x2;
typedef __attribute__((ext_vector_type(8)))  __bf16 bf16x8;
typedef __attribute__((ext_vector_type(16))) __bf16 bf16x16;
typedef __attribute__((ext_vector_type(8)))  float  v8f;

#define D_MODEL 1024
#define FFDIM   4096
#define SEQ     1024
#define BATCH   4
#define NLAYER  6

union FragAB {
  bf16x16 v;
  struct { bf16x8 lo; bf16x8 hi; } h;
};

// ---------------------------------------------------------------------------
// Block reductions (256 threads = 8 wave32)
// ---------------------------------------------------------------------------
__device__ inline float waveReduceSum(float v) {
  #pragma unroll
  for (int o = 16; o > 0; o >>= 1) v += __shfl_xor(v, o, 32);
  return v;
}
__device__ inline float waveReduceMax(float v) {
  #pragma unroll
  for (int o = 16; o > 0; o >>= 1) v = fmaxf(v, __shfl_xor(v, o, 32));
  return v;
}
__device__ inline float blockReduceSum(float v, float* sm) {
  v = waveReduceSum(v);
  if ((threadIdx.x & 31) == 0) sm[threadIdx.x >> 5] = v;
  __syncthreads();
  float r = 0.f;
  #pragma unroll
  for (int i = 0; i < 8; ++i) r += sm[i];
  __syncthreads();
  return r;
}
__device__ inline float blockReduceMax(float v, float* sm) {
  v = waveReduceMax(v);
  if ((threadIdx.x & 31) == 0) sm[threadIdx.x >> 5] = v;
  __syncthreads();
  float r = -3.0e38f;
  #pragma unroll
  for (int i = 0; i < 8; ++i) r = fmaxf(r, sm[i]);
  __syncthreads();
  return r;
}

// ---------------------------------------------------------------------------
// WMMA GEMM: C[M,N] = A[M,K](bf16) * B + bias, epilogue scale/relu,
// optional fp32 and/or bf16 stores. Batched via blockIdx.z.
//   BMODE==0 : B = fp32 weights, row-major [K,N] (NN), converted to bf16 in
//              LDS via packed bf16x2 stores
//   BMODE==1 : B = bf16, row-major [N,K] (NT; used for Q*K^T)
// Block tile 128x128, 8 waves, per-wave 64x32 (4x2 wmma frags).
// LDS K-tile = 64 -> 16 v_wmma per barrier phase.
// Software pipeline: tile kt+64 is fetched into registers while tile kt's
// WMMAs run, so global latency hides under the matrix ops.
// All dims assumed multiples of tile sizes (true for this model).
// ---------------------------------------------------------------------------
template<int BMODE>
__global__ __launch_bounds__(256) void gemm_wmma(
    const bf16* __restrict__ A, long long strideA, int lda,
    const void* __restrict__ Bsrc, long long strideB, int ldb,
    const float* __restrict__ bias,
    float* __restrict__ outF, bf16* __restrict__ outH,
    long long strideC, int ldc, int Kdim, float scale, int relu)
{
  // 72-element (144-byte) row pitch: 16B-aligned rows, staggered banks
  __shared__ __align__(16) bf16 As[128][72];
  __shared__ __align__(16) bf16 Bs[128][72];

  const int t      = threadIdx.x;
  const int lane   = t & 31;
  const int wave   = t >> 5;
  const int laneHi = lane >> 4;   // 0/1: which half-wave
  const int ln     = lane & 15;
  const int waveM  = wave >> 2;   // 0..1
  const int waveN  = wave & 3;    // 0..3
  const int z      = blockIdx.z;
  const int m0     = blockIdx.y * 128;
  const int n0     = blockIdx.x * 128;

  const bf16*  Ab     = A + (long long)z * strideA + (long long)m0 * lda;
  const float* WbBase = (BMODE == 0)
      ? (const float*)Bsrc + (long long)z * strideB + n0 : nullptr;
  const bf16*  BbBase = (BMODE == 1)
      ? (const bf16*)Bsrc + (long long)z * strideB + (long long)n0 * ldb
      : nullptr;

  v8f acc[4][2];
  #pragma unroll
  for (int mi = 0; mi < 4; ++mi)
    #pragma unroll
    for (int ni = 0; ni < 2; ++ni)
      #pragma unroll
      for (int e = 0; e < 8; ++e) acc[mi][ni][e] = 0.f;

  // register staging for the software pipeline
  bf16x8 stgA[4];
  float4 stgB0[8];    // BMODE==0
  bf16x8 stgB1[4];    // BMODE==1

  auto fetch = [&](int kt) {
    #pragma unroll
    for (int i = 0; i < 4; ++i) {
      int c = t + i * 256, row = c >> 3, col = (c & 7) * 8;
      stgA[i] = *(const bf16x8*)(Ab + (long long)row * lda + kt + col);
    }
    if (BMODE == 0) {
      // weight stream: prefetch 2 tiles ahead into GL2
      if (kt + 128 < Kdim)
        __builtin_prefetch((const char*)WbBase +
                           ((size_t)(kt + 128) * ldb) * 4 + t * 64, 0, 1);
      #pragma unroll
      for (int i = 0; i < 4; ++i) {
        int u = t + i * 256;
        int k = (u >> 5) * 2;       // 32 k-pairs x 32 n4-groups
        int n4 = (u & 31) * 4;
        const float* p = WbBase + (long long)(kt + k) * ldb + n4;
        stgB0[2 * i]     = *(const float4*)p;
        stgB0[2 * i + 1] = *(const float4*)(p + ldb);
      }
    } else {
      #pragma unroll
      for (int i = 0; i < 4; ++i) {
        int c = t + i * 256, row = c >> 3, col = (c & 7) * 8;
        stgB1[i] = *(const bf16x8*)(BbBase + (long long)row * ldb + kt + col);
      }
    }
  };

  auto stage_store = [&]() {
    #pragma unroll
    for (int i = 0; i < 4; ++i) {
      int c = t + i * 256, row = c >> 3, col = (c & 7) * 8;
      *(bf16x8*)&As[row][col] = stgA[i];
    }
    if (BMODE == 0) {
      #pragma unroll
      for (int i = 0; i < 4; ++i) {
        int u = t + i * 256;
        int k = (u >> 5) * 2;
        int n4 = (u & 31) * 4;
        float4 w0 = stgB0[2 * i], w1 = stgB0[2 * i + 1];
        bf16x2 p0; p0[0] = (bf16)w0.x; p0[1] = (bf16)w1.x;
        bf16x2 p1; p1[0] = (bf16)w0.y; p1[1] = (bf16)w1.y;
        bf16x2 p2; p2[0] = (bf16)w0.z; p2[1] = (bf16)w1.z;
        bf16x2 p3; p3[0] = (bf16)w0.w; p3[1] = (bf16)w1.w;
        *(bf16x2*)&Bs[n4 + 0][k] = p0;
        *(bf16x2*)&Bs[n4 + 1][k] = p1;
        *(bf16x2*)&Bs[n4 + 2][k] = p2;
        *(bf16x2*)&Bs[n4 + 3][k] = p3;
      }
    } else {
      #pragma unroll
      for (int i = 0; i < 4; ++i) {
        int c = t + i * 256, row = c >> 3, col = (c & 7) * 8;
        *(bf16x8*)&Bs[row][col] = stgB1[i];
      }
    }
  };

  fetch(0);
  for (int kt = 0; kt < Kdim; kt += 64) {
    stage_store();
    __syncthreads();                       // tile visible to all waves
    if (kt + 64 < Kdim) fetch(kt + 64);    // next tile loads fly during WMMAs

    // ---- two K=32 sub-steps per LDS tile; ISA 7.12.2 16-bit layouts ----
    #pragma unroll
    for (int ks = 0; ks < 2; ++ks) {
      const int kb = ks * 32;
      FragAB fa[4], fb[2];
      #pragma unroll
      for (int mi = 0; mi < 4; ++mi) {
        const bf16* rowA = &As[waveM * 64 + mi * 16 + ln][kb];
        // lanes 0-15: K 0-7 & 16-23 ; lanes 16-31: K 8-15 & 24-31
        fa[mi].h.lo = *(const bf16x8*)(rowA + 8 * laneHi);
        fa[mi].h.hi = *(const bf16x8*)(rowA + 8 * laneHi + 16);
      }
      #pragma unroll
      for (int ni = 0; ni < 2; ++ni) {
        const bf16* rowB = &Bs[waveN * 32 + ni * 16 + ln][kb];
        // lanes 0-15: K 0-7 & 16-23 ; lanes 16-31: K 8-15 & 24-31
        fb[ni].h.lo = *(const bf16x8*)(rowB + 8 * laneHi);
        fb[ni].h.hi = *(const bf16x8*)(rowB + 8 * laneHi + 16);
      }
      #pragma unroll
      for (int mi = 0; mi < 4; ++mi)
        #pragma unroll
        for (int ni = 0; ni < 2; ++ni)
          acc[mi][ni] = __builtin_amdgcn_wmma_f32_16x16x32_bf16(
              false, fa[mi].v, false, fb[ni].v,
              (short)0, acc[mi][ni], false, false);
    }
    __syncthreads();                       // all waves done reading LDS
  }

  // ---- epilogue: C lane map M = r + 8*laneHi, N = lane&15 ----
  #pragma unroll
  for (int ni = 0; ni < 2; ++ni) {
    int n = n0 + waveN * 32 + ni * 16 + ln;
    float bvv = bias ? bias[n] : 0.f;
    #pragma unroll
    for (int mi = 0; mi < 4; ++mi) {
      #pragma unroll
      for (int r = 0; r < 8; ++r) {
        int m = m0 + waveM * 64 + mi * 16 + r + 8 * laneHi;
        float v = (acc[mi][ni][r] + bvv) * scale;
        if (relu) v = fmaxf(v, 0.f);
        long long off = (long long)z * strideC + (long long)m * ldc + n;
        if (outF) outF[off] = v;
        if (outH) outH[off] = (bf16)v;
      }
    }
  }
}

// ---------------------------------------------------------------------------
// Embedding + sinusoidal positional encoding. One block per token row.
// ---------------------------------------------------------------------------
__global__ __launch_bounds__(256) void embed_kernel(
    const int* __restrict__ ids, const float* __restrict__ emb,
    float* __restrict__ x, bf16* __restrict__ xb)
{
  int row = blockIdx.x;                 // b*SEQ + s
  int s   = row & (SEQ - 1);
  int id  = ids[row];
  const float* e = emb + (long long)id * D_MODEL;
  #pragma unroll
  for (int i = 0; i < 4; ++i) {
    int d  = threadIdx.x + i * 256;
    int j2 = d & ~1;                    // even index 2*(d/2)
    float freq = expf((float)j2 * (-9.210340371976184f / (float)D_MODEL));
    float ang  = (float)s * freq;
    float pe   = (d & 1) ? cosf(ang) : sinf(ang);
    float v    = e[d] + pe;
    long long off = (long long)row * D_MODEL + d;
    x[off]  = v;
    xb[off] = (bf16)v;
  }
}

// ---------------------------------------------------------------------------
// x = LayerNorm(xin + delta) * g + beta. One block per row (in-place safe).
// ---------------------------------------------------------------------------
__global__ __launch_bounds__(256) void add_ln_kernel(
    const float* __restrict__ xin, const float* __restrict__ delta,
    const float* __restrict__ g, const float* __restrict__ beta,
    float* __restrict__ xout, bf16* __restrict__ xbout)
{
  __shared__ float sm[8];
  int row = blockIdx.x;
  long long base = (long long)row * D_MODEL;
  float vals[4];
  float s = 0.f, s2 = 0.f;
  #pragma unroll
  for (int i = 0; i < 4; ++i) {
    int d = threadIdx.x + i * 256;
    float v = xin[base + d] + delta[base + d];
    vals[i] = v;
    s += v; s2 += v * v;
  }
  float ts  = blockReduceSum(s, sm);
  float ts2 = blockReduceSum(s2, sm);
  float mu  = ts * (1.f / D_MODEL);
  float var = ts2 * (1.f / D_MODEL) - mu * mu;
  float inv = rsqrtf(var + 1e-5f);
  #pragma unroll
  for (int i = 0; i < 4; ++i) {
    int d = threadIdx.x + i * 256;
    float y = (vals[i] - mu) * inv * g[d] + beta[d];
    xout[base + d]  = y;
    xbout[base + d] = (bf16)y;
  }
}

// ---------------------------------------------------------------------------
// Masked softmax over last dim; fp32 scores in, bf16 probs out.
// ---------------------------------------------------------------------------
__global__ __launch_bounds__(256) void softmax_kernel(
    const float* __restrict__ scores, const int* __restrict__ amask,
    bf16* __restrict__ attn)
{
  __shared__ float sm[8];
  int row = blockIdx.x;                 // b*SEQ + q
  int b   = row >> 10;                  // SEQ == 1024
  long long base = (long long)row * SEQ;
  float v[4];
  float mx = -3.0e38f;
  #pragma unroll
  for (int i = 0; i < 4; ++i) {
    int k = threadIdx.x + i * 256;
    float sc = scores[base + k];
    sc = amask[b * SEQ + k] ? sc : -1.0e9f;
    v[i] = sc;
    mx = fmaxf(mx, sc);
  }
  mx = blockReduceMax(mx, sm);
  float sum = 0.f;
  #pragma unroll
  for (int i = 0; i < 4; ++i) { v[i] = expf(v[i] - mx); sum += v[i]; }
  sum = blockReduceSum(sum, sm);
  float inv = 1.f / sum;
  #pragma unroll
  for (int i = 0; i < 4; ++i) {
    int k = threadIdx.x + i * 256;
    attn[base + k] = (bf16)(v[i] * inv);
  }
}

// ---------------------------------------------------------------------------
// Host orchestration
// ---------------------------------------------------------------------------
extern "C" void kernel_launch(void* const* d_in, const int* in_sizes, int n_in,
                              void* d_out, int out_size, void* d_ws, size_t ws_size,
                              hipStream_t stream) {
  const int*   ids   = (const int*)d_in[0];
  const int*   amask = (const int*)d_in[1];
  const float* emb   = (const float*)d_in[2];
  const float* Wq    = (const float*)d_in[3];
  const float* Wk    = (const float*)d_in[4];
  const float* Wv    = (const float*)d_in[5];
  const float* bq    = (const float*)d_in[6];
  const float* bk    = (const float*)d_in[7];
  const float* bv    = (const float*)d_in[8];
  const float* g1    = (const float*)d_in[9];
  const float* be1   = (const float*)d_in[10];
  const float* Wf1   = (const float*)d_in[11];
  const float* bf1   = (const float*)d_in[12];
  const float* Wf2   = (const float*)d_in[13];
  const float* bf2   = (const float*)d_in[14];
  const float* g2    = (const float*)d_in[15];
  const float* be2   = (const float*)d_in[16];

  float* x  = (float*)d_out;            // residual stream, [B,S,D] fp32
  char*  ws = (char*)d_ws;
  const size_t MB = 1ull << 20;
  bf16*  xb     = (bf16*)(ws + 0);      //  8 MB  bf16 copy of x
  bf16*  Qb     = (bf16*)(ws + 8*MB);   //  8 MB
  bf16*  Kb     = (bf16*)(ws + 16*MB);  //  8 MB
  float* Vf     = (float*)(ws + 24*MB); // 16 MB
  float* delta  = (float*)(ws + 40*MB); // 16 MB (attn out / ffn out)
  float* scores = (float*)(ws + 56*MB); // 16 MB
  bf16*  attn   = (bf16*)(ws + 72*MB);  //  8 MB
  bf16*  h      = (bf16*)(ws + 56*MB);  // 32 MB, reuses scores+attn (dead by then)

  const int BS = BATCH * SEQ;           // 4096 token rows
  const long long SD = (long long)SEQ * D_MODEL;
  const long long SS = (long long)SEQ * SEQ;
  dim3 blk(256);

  embed_kernel<<<BS, blk, 0, stream>>>(ids, emb, x, xb);

  for (int l = 0; l < NLAYER; ++l) {
    const float* wq  = Wq  + (size_t)l * D_MODEL * D_MODEL;
    const float* wk  = Wk  + (size_t)l * D_MODEL * D_MODEL;
    const float* wv  = Wv  + (size_t)l * D_MODEL * D_MODEL;
    const float* wf1 = Wf1 + (size_t)l * D_MODEL * FFDIM;
    const float* wf2 = Wf2 + (size_t)l * FFDIM * D_MODEL;

    // Q (scale 1/sqrt(D)=1/32 folded in), K, V projections
    gemm_wmma<0><<<dim3(8, 32, 1), blk, 0, stream>>>(
        xb, 0, D_MODEL, wq, 0, D_MODEL, bq + l * D_MODEL,
        nullptr, Qb, 0, D_MODEL, D_MODEL, 0.03125f, 0);
    gemm_wmma<0><<<dim3(8, 32, 1), blk, 0, stream>>>(
        xb, 0, D_MODEL, wk, 0, D_MODEL, bk + l * D_MODEL,
        nullptr, Kb, 0, D_MODEL, D_MODEL, 1.f, 0);
    gemm_wmma<0><<<dim3(8, 32, 1), blk, 0, stream>>>(
        xb, 0, D_MODEL, wv, 0, D_MODEL, bv + l * D_MODEL,
        Vf, nullptr, 0, D_MODEL, D_MODEL, 1.f, 0);

    // scores = Q * K^T  (NT, batched over B)
    gemm_wmma<1><<<dim3(8, 8, BATCH), blk, 0, stream>>>(
        Qb, SD, D_MODEL, Kb, SD, D_MODEL, nullptr,
        scores, nullptr, SS, SEQ, D_MODEL, 1.f, 0);

    softmax_kernel<<<BS, blk, 0, stream>>>(scores, amask, attn);

    // delta = attn * V  (NN, batched over B)
    gemm_wmma<0><<<dim3(8, 8, BATCH), blk, 0, stream>>>(
        attn, SS, SEQ, Vf, SD, D_MODEL, nullptr,
        delta, nullptr, SD, D_MODEL, SEQ, 1.f, 0);

    add_ln_kernel<<<BS, blk, 0, stream>>>(
        x, delta, g1 + l * D_MODEL, be1 + l * D_MODEL, x, xb);

    // h = relu(x * Wf1 + bf1)
    gemm_wmma<0><<<dim3(32, 32, 1), blk, 0, stream>>>(
        xb, 0, D_MODEL, wf1, 0, FFDIM, bf1 + l * FFDIM,
        nullptr, h, 0, FFDIM, D_MODEL, 1.f, 1);
    // delta = h * Wf2 + bf2
    gemm_wmma<0><<<dim3(8, 32, 1), blk, 0, stream>>>(
        h, 0, FFDIM, wf2, 0, D_MODEL, bf2 + l * D_MODEL,
        delta, nullptr, 0, D_MODEL, FFDIM, 1.f, 0);

    add_ln_kernel<<<BS, blk, 0, stream>>>(
        x, delta, g2 + l * D_MODEL, be2 + l * D_MODEL, x, xb);
  }
  (void)in_sizes; (void)n_in; (void)out_size; (void)ws_size;
}
